// NonLocalBlockND_88751204205091
// MI455X (gfx1250) — compile-verified, matrix operand order checked by reading, not run on hardware
//
#include <hip/hip_runtime.h>
#include <hip/hip_bf16.h>

// Problem constants (match reference)
#define B_   8
#define C_   256
#define CI_  128
#define H_   64
#define W_   64
#define N_   (H_*W_)          // 4096 pixels
#define M_   ((H_/2)*(W_/2))  // 1024 pooled pixels
#define EPS_ 1e-5f

typedef __bf16 bf16;
typedef __attribute__((ext_vector_type(16))) __bf16 bf16x16;
typedef __attribute__((ext_vector_type(8)))  __bf16 bf16x8;
typedef __attribute__((ext_vector_type(8)))  float  f32x8;
typedef __attribute__((ext_vector_type(4)))  float  f32x4;

// Fragment element e of a lane (hi = lane>=16) holds K = (e/8)*16 + hi*8 + (e%8).
__device__ __forceinline__ int frag_k(int e, int hi) {
  return ((e >> 3) << 4) | (hi << 3) | (e & 7);
}

// Generic strided loader (for K-major data, e.g. pixel-major x): 16 scalar loads.
__device__ __forceinline__ bf16x16 load_frag_f32_strided(const float* base, int kstride, int hi) {
  bf16x16 f;
#pragma unroll
  for (int e = 0; e < 16; ++e)
    f[e] = (__bf16)base[(size_t)frag_k(e, hi) * kstride];
  return f;
}

// K-contiguous bf16 loader: two 16B vector loads per lane. base must be 16B aligned.
__device__ __forceinline__ bf16x16 load_frag_bf16_k1(const bf16* base, int hi) {
  bf16x8 a0 = *(const bf16x8*)(base + (hi << 3));        // K = hi*8 + 0..7
  bf16x8 a1 = *(const bf16x8*)(base + 16 + (hi << 3));   // K = 16 + hi*8 + 0..7
  bf16x16 f;
#pragma unroll
  for (int e = 0; e < 8; ++e) { f[e] = a0[e]; f[e + 8] = a1[e]; }
  return f;
}

// K-contiguous f32 loader: four 16B vector loads per lane + cvt.
__device__ __forceinline__ bf16x16 load_frag_f32_k1(const float* base, int hi) {
  const float* b0 = base + (hi << 3);
  f32x4 a0 = *(const f32x4*)(b0);
  f32x4 a1 = *(const f32x4*)(b0 + 4);
  f32x4 a2 = *(const f32x4*)(b0 + 16);
  f32x4 a3 = *(const f32x4*)(b0 + 20);
  bf16x16 f;
#pragma unroll
  for (int e = 0; e < 4; ++e) {
    f[e]      = (__bf16)a0[e];
    f[e + 4]  = (__bf16)a1[e];
    f[e + 8]  = (__bf16)a2[e];
    f[e + 12] = (__bf16)a3[e];
  }
  return f;
}

#define WMMA_BF16(acc, a, b)                                                       \
  acc = __builtin_amdgcn_wmma_f32_16x16x32_bf16(false, (a), false, (b), (short)0,  \
                                                (acc), false, false)

// ---------------------------------------------------------------------------
// Kernel 0: fold BN + final-conv bias into per-channel scale/shift.
//   scale[c] = gamma[c] * rsqrt(var[c]+eps)
//   shift[c] = beta[c] + (b_w[c] - mean[c]) * scale[c]
// ---------------------------------------------------------------------------
__global__ __launch_bounds__(C_) void k_bnfold(
    const float* __restrict__ b_w, const float* __restrict__ gamma,
    const float* __restrict__ beta, const float* __restrict__ mean,
    const float* __restrict__ var,
    float* __restrict__ scale, float* __restrict__ shift)
{
  const int c = threadIdx.x;
  const float sc = gamma[c] * rsqrtf(var[c] + EPS_);
  scale[c] = sc;
  shift[c] = beta[c] + (b_w[c] - mean[c]) * sc;
}

// ---------------------------------------------------------------------------
// Kernel 1: fused theta/phi/g 1x1-conv (+bias) with 2x2 max-pool for phi/g.
// Block = 256 thr (8 waves): one 16-channel tile x one 2-image-row strip
// (128 contiguous pixels). Wave w handles pixels [strip*128 + w*16, +16).
// theta stored TRANSPOSED bf16 [B,N,CI] (one 16B store/lane; K-contiguous for
// the consumer); phi/g pooled through LDS, stored bf16 [B,CI,M].
// ---------------------------------------------------------------------------
__global__ __launch_bounds__(256) void k_conv3(
    const float* __restrict__ x,
    const float* __restrict__ w_theta, const float* __restrict__ b_theta,
    const float* __restrict__ w_phi,   const float* __restrict__ b_phi,
    const float* __restrict__ w_g,     const float* __restrict__ b_g,
    bf16* __restrict__ thetaT_o, bf16* __restrict__ phi_o, bf16* __restrict__ g_o)
{
  __shared__ float lds_p[16][128];
  __shared__ float lds_g[16][128];

  const int b     = blockIdx.y;
  const int ct    = blockIdx.x & 7;    // channel tile (CI/16 = 8)
  const int strip = blockIdx.x >> 3;   // image row pair (H/2 = 32)
  const int wv    = threadIdx.x >> 5;
  const int lane  = threadIdx.x & 31;
  const int hi    = lane >> 4;
  const int lo    = lane & 15;

  const int oc0 = ct * 16;
  const int n0  = strip * 128 + wv * 16;

  const float* xb = x + (size_t)b * C_ * N_;

  f32x8 acc_t = {}; f32x8 acc_p = {}; f32x8 acc_g = {};
#pragma unroll
  for (int kk = 0; kk < C_; kk += 32) {
    // B operand: x as [K=C, N] pixel-major -> strided
    bf16x16 bx = load_frag_f32_strided(xb + (size_t)kk * N_ + n0 + lo, N_, hi);
    if (kk + 32 < C_)  // uniform branch; prefetch next K-slab of x
      __builtin_prefetch(xb + (size_t)(kk + 32) * N_ + n0 + lo, 0, 1);
    // A operands: weights [CI, C] row-major -> K-contiguous
    bf16x16 at = load_frag_f32_k1(w_theta + (oc0 + lo) * C_ + kk, hi);
    bf16x16 ap = load_frag_f32_k1(w_phi   + (oc0 + lo) * C_ + kk, hi);
    bf16x16 ag = load_frag_f32_k1(w_g     + (oc0 + lo) * C_ + kk, hi);
    WMMA_BF16(acc_t, at, bx);
    WMMA_BF16(acc_p, ap, bx);
    WMMA_BF16(acc_g, ag, bx);
  }

  // D VGPR v <-> channel c0 + v (8 consecutive channels per lane half).
  const int c0 = oc0 + (hi << 3);
  const int n  = n0 + lo;
  {
    f32x4 bt0 = *(const f32x4*)(b_theta + c0);
    f32x4 bt1 = *(const f32x4*)(b_theta + c0 + 4);
    f32x4 bp0 = *(const f32x4*)(b_phi + c0);
    f32x4 bp1 = *(const f32x4*)(b_phi + c0 + 4);
    f32x4 bg0 = *(const f32x4*)(b_g + c0);
    f32x4 bg1 = *(const f32x4*)(b_g + c0 + 4);
    bf16x8 tv;
#pragma unroll
    for (int v = 0; v < 8; ++v) {
      const float bt = (v < 4) ? bt0[v] : bt1[v - 4];
      const float bp = (v < 4) ? bp0[v] : bp1[v - 4];
      const float bg = (v < 4) ? bg0[v] : bg1[v - 4];
      tv[v] = (__bf16)(acc_t[v] + bt);
      lds_p[(hi << 3) + v][wv * 16 + lo] = acc_p[v] + bp;
      lds_g[(hi << 3) + v][wv * 16 + lo] = acc_g[v] + bg;
    }
    // theta^T [B, N, CI]: one aligned 16B store per lane
    *(bf16x8*)(thetaT_o + (size_t)b * N_ * CI_ + (size_t)n * CI_ + c0) = tv;
  }
  __syncthreads();

  // 2x2 max pool: strip = rows (2*strip, 2*strip+1); LDS px [0,64)=row0, [64,128)=row1
  for (int idx = threadIdx.x; idx < 16 * 32; idx += 256) {
    const int chl = idx >> 5;
    const int pc  = idx & 31;
    const int cc  = 2 * pc;
    const float pv = fmaxf(fmaxf(lds_p[chl][cc],      lds_p[chl][cc + 1]),
                           fmaxf(lds_p[chl][64 + cc], lds_p[chl][64 + cc + 1]));
    const float gv = fmaxf(fmaxf(lds_g[chl][cc],      lds_g[chl][cc + 1]),
                           fmaxf(lds_g[chl][64 + cc], lds_g[chl][64 + cc + 1]));
    const int ch = oc0 + chl;
    const int pm = strip * 32 + pc;  // pooled pixel index
    phi_o[(size_t)b * CI_ * M_ + (size_t)ch * M_ + pm] = (bf16)pv;
    g_o  [(size_t)b * CI_ * M_ + (size_t)ch * M_ + pm] = (bf16)gv;
  }
}

// ---------------------------------------------------------------------------
// Kernel 2: S[b] = phi[b] @ g[b]^T  -> [CI, CI] f32, K = M_ (1024)
// S[ci, cj] = sum_m phi[ci,m] * g[cj,m]   (both operands K-contiguous)
// ---------------------------------------------------------------------------
__global__ __launch_bounds__(256) void k_gram(
    const bf16* __restrict__ phi, const bf16* __restrict__ g, float* __restrict__ S)
{
  const int b    = blockIdx.x >> 3;
  const int wv   = threadIdx.x >> 5;
  const int lane = threadIdx.x & 31;
  const int hi = lane >> 4, lo = lane & 15;
  const int tile = ((blockIdx.x & 7) << 3) + wv;   // 0..63
  const int ti = tile >> 3, tj = tile & 7;

  const bf16* pb = phi + (size_t)b * CI_ * M_ + (size_t)(ti * 16 + lo) * M_;
  const bf16* gb = g   + (size_t)b * CI_ * M_ + (size_t)(tj * 16 + lo) * M_;

  f32x8 acc = {};
  for (int kk = 0; kk < M_; kk += 32) {
    bf16x16 a  = load_frag_bf16_k1(pb + kk, hi);
    bf16x16 bb = load_frag_bf16_k1(gb + kk, hi);
    WMMA_BF16(acc, a, bb);
  }
#pragma unroll
  for (int v = 0; v < 8; ++v) {
    const int ci = ti * 16 + v + hi * 8;
    const int cj = tj * 16 + lo;
    S[(size_t)b * CI_ * CI_ + ci * CI_ + cj] = acc[v];
  }
}

// ---------------------------------------------------------------------------
// Kernel 3: A_eff[b] = w_w @ S[b]^T / M_  -> [C, CI] bf16, K = CI (cj)
// A_eff[c, ci] = (1/M) * sum_cj w_w[c, cj] * S[ci, cj]
// ---------------------------------------------------------------------------
__global__ __launch_bounds__(256) void k_amat(
    const float* __restrict__ w_w, const float* __restrict__ S, bf16* __restrict__ A)
{
  const int b    = blockIdx.x >> 4;
  const int wv   = threadIdx.x >> 5;
  const int lane = threadIdx.x & 31;
  const int hi = lane >> 4, lo = lane & 15;
  const int tile = ((blockIdx.x & 15) << 3) + wv;  // 0..127
  const int ti = tile >> 3;  // C tile 0..15
  const int tj = tile & 7;   // CI tile 0..7

  const float* Sb = S + (size_t)b * CI_ * CI_;

  f32x8 acc = {};
#pragma unroll
  for (int kk = 0; kk < CI_; kk += 32) {
    bf16x16 a  = load_frag_f32_k1(w_w + (size_t)(ti * 16 + lo) * CI_ + kk, hi);
    // B[k=cj, n=ci] = S[ci*CI + cj] -> K-contiguous per row ci
    bf16x16 bb = load_frag_f32_k1(Sb + (size_t)(tj * 16 + lo) * CI_ + kk, hi);
    WMMA_BF16(acc, a, bb);
  }
  const float invM = 1.0f / (float)M_;
#pragma unroll
  for (int v = 0; v < 8; ++v) {
    const int c  = ti * 16 + v + hi * 8;
    const int ci = tj * 16 + lo;
    A[(size_t)b * C_ * CI_ + c * CI_ + ci] = (bf16)(acc[v] * invM);
  }
}

// ---------------------------------------------------------------------------
// Kernel 4: out = scale_c*(A_eff @ theta) + shift_c + x   (BN + bias + residual)
// [C, N] per batch, K = CI (4 WMMA steps). theta^T makes B K-contiguous.
// ---------------------------------------------------------------------------
__global__ __launch_bounds__(256) void k_out(
    const bf16* __restrict__ Amat, const bf16* __restrict__ thetaT,
    const float* __restrict__ x,
    const float* __restrict__ scale, const float* __restrict__ shift,
    float* __restrict__ out)
{
  const int b    = blockIdx.y;
  const int wv   = threadIdx.x >> 5;
  const int lane = threadIdx.x & 31;
  const int hi = lane >> 4, lo = lane & 15;
  const int tile = blockIdx.x * 8 + wv;  // 0..4095
  const int ti = tile >> 8;    // C tile 0..15
  const int tj = tile & 255;   // N tile 0..255

  const int n = tj * 16 + lo;
  const bf16* Ab = Amat   + (size_t)b * C_ * CI_ + (size_t)(ti * 16 + lo) * CI_;
  const bf16* tb = thetaT + (size_t)b * N_ * CI_ + (size_t)n * CI_;

  f32x8 acc = {};
#pragma unroll
  for (int kk = 0; kk < CI_; kk += 32) {
    bf16x16 a  = load_frag_bf16_k1(Ab + kk, hi);
    // B[k=ci, n] = thetaT[n*CI + ci] -> K-contiguous
    bf16x16 bb = load_frag_bf16_k1(tb + kk, hi);
    WMMA_BF16(acc, a, bb);
  }

  const int c0 = ti * 16 + (hi << 3);   // 8 consecutive channels per lane half
  f32x4 sc0 = *(const f32x4*)(scale + c0);
  f32x4 sc1 = *(const f32x4*)(scale + c0 + 4);
  f32x4 sh0 = *(const f32x4*)(shift + c0);
  f32x4 sh1 = *(const f32x4*)(shift + c0 + 4);
  const size_t base = (size_t)b * C_ * N_ + (size_t)c0 * N_ + n;
  const float* xp = x + base;
  float* op = out + base;
#pragma unroll
  for (int v = 0; v < 8; ++v) {
    const float sc = (v < 4) ? sc0[v] : sc1[v - 4];
    const float sh = (v < 4) ? sh0[v] : sh1[v - 4];
    op[(size_t)v * N_] = acc[v] * sc + sh + xp[(size_t)v * N_];
  }
}

// ---------------------------------------------------------------------------
extern "C" void kernel_launch(void* const* d_in, const int* in_sizes, int n_in,
                              void* d_out, int out_size, void* d_ws, size_t ws_size,
                              hipStream_t stream) {
  (void)in_sizes; (void)n_in; (void)out_size; (void)ws_size;
  const float* x       = (const float*)d_in[0];
  const float* w_theta = (const float*)d_in[1];
  const float* b_theta = (const float*)d_in[2];
  const float* w_phi   = (const float*)d_in[3];
  const float* b_phi   = (const float*)d_in[4];
  const float* w_g     = (const float*)d_in[5];
  const float* b_g     = (const float*)d_in[6];
  const float* w_w     = (const float*)d_in[7];
  const float* b_w     = (const float*)d_in[8];
  const float* gamma   = (const float*)d_in[9];
  const float* beta    = (const float*)d_in[10];
  const float* mean    = (const float*)d_in[11];
  const float* var     = (const float*)d_in[12];
  float* out = (float*)d_out;

  char* ws = (char*)d_ws;
  bf16*  thetaT_ws = (bf16*) (ws);                               // 8 MiB  [B,N,CI]
  bf16*  phi_ws    = (bf16*) (ws + (8ull  << 20));               // 2 MiB  [B,CI,M]
  bf16*  g_ws      = (bf16*) (ws + (10ull << 20));               // 2 MiB  [B,CI,M]
  float* S_ws      = (float*)(ws + (12ull << 20));               // 512 KiB [B,CI,CI]
  bf16*  A_ws      = (bf16*) (ws + (12ull << 20) + (1u << 19));  // 512 KiB [B,C,CI]
  float* scale_ws  = (float*)(ws + (13ull << 20));               // 1 KiB  [C]
  float* shift_ws  = (float*)(ws + (13ull << 20) + 4096);        // 1 KiB  [C]

  // 0) fold BN params (+ final-conv bias) into per-channel scale/shift
  k_bnfold<<<dim3(1), C_, 0, stream>>>(b_w, gamma, beta, mean, var, scale_ws, shift_ws);
  // 1) theta/phi/g convs + pool.  grid: (8 ch-tiles * 32 strips) x B
  k_conv3<<<dim3(8 * 32, B_), 256, 0, stream>>>(x, w_theta, b_theta, w_phi, b_phi,
                                                w_g, b_g, thetaT_ws, phi_ws, g_ws);
  // 2) S = phi g^T.  64 tiles/batch, 8 waves/block -> 8 blocks/batch
  k_gram<<<dim3(B_ * 8), 256, 0, stream>>>(phi_ws, g_ws, S_ws);
  // 3) A_eff = w_w S^T / M.  128 tiles/batch -> 16 blocks/batch
  k_amat<<<dim3(B_ * 16), 256, 0, stream>>>(w_w, S_ws, A_ws);
  // 4) out = scale*(A_eff theta) + shift + x.  4096 tiles/batch -> 512 blocks x B
  k_out<<<dim3(512, B_), 256, 0, stream>>>(A_ws, thetaT_ws, x, scale_ws, shift_ws, out);
}
